// MultiheadAttention_58299886076305
// MI455X (gfx1250) — matrix operand order repaired and checked by reading, not compiled, not run on hardware
//
#include <hip/hip_runtime.h>
#include <math.h>

// Problem constants (reference: E=1024, H=16, KD=VD=64, B=4, S=2048)
#define EDIM 1024
#define HDIM 16
#define DK   64
#define DV   64
#define BDIM 4
#define SDIM 2048
#define BS   (BDIM * SDIM)          // 8192 rows
#define HV   (HDIM * DV)            // 1024

typedef __attribute__((ext_vector_type(16))) _Float16 v16h;
typedef __attribute__((ext_vector_type(8)))  _Float16 v8h;
typedef __attribute__((ext_vector_type(4)))  _Float16 v4h;
typedef __attribute__((ext_vector_type(8)))  float    v8f;
typedef __attribute__((ext_vector_type(4)))  float    v4f;
typedef __attribute__((ext_vector_type(4)))  unsigned int v4u;
typedef __attribute__((ext_vector_type(8)))  int      v8i;
typedef __attribute__((ext_vector_type(4)))  int      v4i;

__device__ __forceinline__ v8f wmma_f16(v16h a, v16h b, v8f c) {
  return __builtin_amdgcn_wmma_f32_16x16x32_f16(
      false, a, false, b, (short)0, c, false, false);
}

__device__ __forceinline__ v16h cat8(v8h lo, v8h hi) {
  return __builtin_shufflevector(lo, hi, 0,1,2,3,4,5,6,7,8,9,10,11,12,13,14,15);
}

// ---- fragment loaders: all per-lane CONTIGUOUS -> ds_load_b128 ----
// A 16x32 (MxK), source row-major [row][k], ld multiple of 8 halfs, k0 mult of 8.
__device__ __forceinline__ v16h load_a_frag(const _Float16* p, int row0, int k0, int ld) {
  const int lane = threadIdx.x & 31;
  const int m  = lane & 15;
  const int kh = (lane >> 4) * 8;
  const _Float16* r = p + (size_t)(row0 + m) * ld + k0;
  v8h lo = *(const v8h*)(r + kh);
  v8h hi = *(const v8h*)(r + 16 + kh);
  return cat8(lo, hi);
}

// B 32x16 (KxN) where the SOURCE is stored transposed [n][k] row-major.
__device__ __forceinline__ v16h load_bt_frag(const _Float16* p, int n0, int k0, int ld) {
  const int lane = threadIdx.x & 31;
  const int n  = lane & 15;
  const int kb = (lane >> 4) * 8;
  const _Float16* r = p + (size_t)(n0 + n) * ld + k0;
  v8h lo = *(const v8h*)(r + kb);
  v8h hi = *(const v8h*)(r + 16 + kb);
  return cat8(lo, hi);
}

// ---- Tensor Data Mover: async 2D f16 tile -> LDS (DMA writes row padding) ----
__device__ __forceinline__ void tdm_load_2d_f16(
    unsigned lds_off_bytes, const void* gaddr,
    unsigned tensor_d0, unsigned tensor_d1, unsigned stride0,
    unsigned tile_d0, unsigned tile_d1,
    int pad_interval_code, int pad_amount_code) {
  unsigned long long ga = (unsigned long long)(uintptr_t)gaddr;
  v4u g0;
  g0[0] = 1u;                                                // count=1
  g0[1] = lds_off_bytes;                                     // lds_addr
  g0[2] = (unsigned)ga;                                      // global_addr[31:0]
  g0[3] = (unsigned)((ga >> 32) & 0x01FFFFFFu) | (2u << 30); // addr[56:32]|type=2
  v8i g1;
  g1[0] = (1 << 16)                                          // data_size = 2B
        | (1 << 20)                                          // pad_enable
        | (pad_interval_code << 22) | (pad_amount_code << 25);
  g1[1] = (int)((tensor_d0 & 0xffffu) << 16);
  g1[2] = (int)(((tensor_d0 >> 16) & 0xffffu) | ((tensor_d1 & 0xffffu) << 16));
  g1[3] = (int)(((tensor_d1 >> 16) & 0xffffu) | ((tile_d0 & 0xffffu) << 16));
  g1[4] = (int)(tile_d1 & 0xffffu);
  g1[5] = (int)stride0;
  g1[6] = 0;
  g1[7] = 0;
  v4i z4 = {0, 0, 0, 0};
#if __has_include(<hip/amd_detail/amd_gfx1250_TDM.h>)
  v8i z8 = {0, 0, 0, 0, 0, 0, 0, 0};
  __builtin_amdgcn_tensor_load_to_lds(g0, g1, z4, z4, z8, 0);
#else
  __builtin_amdgcn_tensor_load_to_lds(g0, g1, z4, z4, 0);
#endif
}

__device__ __forceinline__ unsigned lds_offset(const void* p) {
  return (unsigned)(uintptr_t)p;   // generic LDS addr[31:0] == LDS byte offset
}

// =====================================================================
// Kernel 1: fused QKV projection.  grid = (BS/64, 3*H), block = 128.
// K-chunk = 64: 8 WMMAs per barrier.
// =====================================================================
__global__ __launch_bounds__(128) void qkv_proj_kernel(
    const float* __restrict__ X,
    const float* __restrict__ W_Q, const float* __restrict__ W_K, const float* __restrict__ W_V,
    const float* __restrict__ B_Q, const float* __restrict__ B_K, const float* __restrict__ B_V,
    _Float16* __restrict__ Qh, _Float16* __restrict__ Kh, _Float16* __restrict__ Vh)
{
  const int mtile = blockIdx.x;
  const int which = blockIdx.y / HDIM;
  const int h     = blockIdx.y % HDIM;
  const float* W  = (which == 0) ? W_Q : (which == 1) ? W_K : W_V;
  const float* Bb = (which == 0) ? B_Q : (which == 1) ? B_K : B_V;
  _Float16*  Out  = (which == 0) ? Qh  : (which == 1) ? Kh  : Vh;

  const int row0 = mtile * 64;
  const int wave = threadIdx.x >> 5;
  const int lane = threadIdx.x & 31;

  __shared__ __align__(16) _Float16 lA [64][72];  // [m][k]  64x64 (+pad)
  __shared__ __align__(16) _Float16 lBt[64][72];  // [n][k]  64x64 transposed W

  v8f acc[4] = {};

  for (int kc = 0; kc < EDIM; kc += 64) {
    // X tile: float4 global reads, b64 LDS stores
    for (int i = threadIdx.x; i < (64 * 64) / 4; i += 128) {
      int r = i >> 4, c = (i & 15) * 4;
      v4f x = *(const v4f*)&X[(size_t)(row0 + r) * EDIM + kc + c];
      v4h hx;
#pragma unroll
      for (int j = 0; j < 4; ++j) hx[j] = (_Float16)x[j];
      *(v4h*)&lA[r][c] = hx;
    }
    // W chunk: coalesced float4 along n, stored transposed [n][k]
    for (int i = threadIdx.x; i < (64 * 64) / 4; i += 128) {
      int k = i >> 4, n4 = (i & 15) * 4;
      v4f w = *(const v4f*)&W[((size_t)h * EDIM + kc + k) * DK + n4];
#pragma unroll
      for (int j = 0; j < 4; ++j) lBt[n4 + j][k] = (_Float16)w[j];
    }
    __syncthreads();

    v16h a0 = load_a_frag(&lA[0][0], wave * 16, 0,  72);
    v16h a1 = load_a_frag(&lA[0][0], wave * 16, 32, 72);
#pragma unroll
    for (int nt = 0; nt < 4; ++nt) {
      acc[nt] = wmma_f16(a0, load_bt_frag(&lBt[0][0], nt * 16, 0,  72), acc[nt]);
      acc[nt] = wmma_f16(a1, load_bt_frag(&lBt[0][0], nt * 16, 32, 72), acc[nt]);
    }
    __syncthreads();
  }

  const int hlf = lane >> 4, nl = lane & 15;
  const int bidx   = row0 / SDIM;
  const int s_base = (row0 % SDIM) + wave * 16;
#pragma unroll
  for (int nt = 0; nt < 4; ++nt) {
#pragma unroll
    for (int v = 0; v < 8; ++v) {
      int m = v + 8 * hlf;
      int n = nt * 16 + nl;
      float val = acc[nt][v] + Bb[h * DK + n];
      size_t idx = (((size_t)bidx * HDIM + h) * SDIM + (s_base + m)) * DK + n;
      Out[idx] = (_Float16)val;
    }
  }
}

// =====================================================================
// Kernel 2: causal flash attention.  grid = (S/64, H, B), block = 128.
// KV tile = 64: 16 WMMAs per stage. K streamed by TDM (double-buffered,
// DMA row-padded); V staged transposed for contiguous B-fragments.
// =====================================================================
__global__ __launch_bounds__(128) void flash_attn_kernel(
    const _Float16* __restrict__ Qh, const _Float16* __restrict__ Kh,
    const _Float16* __restrict__ Vh, _Float16* __restrict__ AO)
{
  const int qt = blockIdx.x, h = blockIdx.y, b = blockIdx.z;
  const int wave = threadIdx.x >> 5;
  const int lane = threadIdx.x & 31;
  const int hlf = lane >> 4, nl = lane & 15;

  const int q0_wg = qt * 64;
  const int q0    = q0_wg + wave * 16;

  const size_t head_off = ((size_t)b * HDIM + h) * (size_t)SDIM * DK;
  const _Float16* Qp = Qh + head_off;
  const _Float16* Kp = Kh + head_off;
  const _Float16* Vp = Vh + head_off;

  __shared__ __align__(16) _Float16 lK [2][64][72];   // [s][kd] double-buffer (TDM)
  __shared__ __align__(16) _Float16 lVt[64][72];      // [dv][s] transposed
  __shared__ __align__(16) _Float16 lP [4][16][72];   // per-wave P patch (16x64)

  // Q fragments (A layout): contiguous b128 reads straight from global f16
  v16h qa[2];
  {
    const int m = nl, kh = hlf * 8;
#pragma unroll
    for (int f = 0; f < 2; ++f) {
      const _Float16* qrow = Qp + (size_t)(q0 + m) * DK + f * 32;
      qa[f] = cat8(*(const v8h*)(qrow + kh), *(const v8h*)(qrow + 16 + kh));
    }
  }

  float mrow[8], lrow[8];
  v8f acc[4] = {};
#pragma unroll
  for (int v = 0; v < 8; ++v) { mrow[v] = -INFINITY; lrow[v] = 0.f; }

  const float scale = 0.125f;              // 1/sqrt(64)
  const int  nkv    = q0_wg / 64 + 1;      // 64-wide causal tiles
  const bool issuer = (threadIdx.x < 32);  // wave 0 drives the TDM

  // prime: DMA K tile 0 (64 rows x 64 halfs = 32 dwords/row, +4dw pad -> ld 72)
  if (issuer)
    tdm_load_2d_f16(lds_offset(&lK[0][0][0]), Kp, DK, 64, DK, DK, 64,
                    /*interval 32dw=*/4, /*pad 4dw=*/3);

  for (int t = 0; t < nkv; ++t) {
    const int kv0 = t * 64;
    const int buf = t & 1;

    // stage V tile transposed: v8h global reads, scalar transpose stores
    for (int i = threadIdx.x; i < (64 * 64) / 8; i += 128) {
      int r = i >> 3, c = (i & 7) * 8;
      v8h v = *(const v8h*)&Vp[(size_t)(kv0 + r) * DK + c];
#pragma unroll
      for (int j = 0; j < 8; ++j) lVt[c + j][r] = v[j];
    }

    if (issuer) {
      if (t + 1 < nkv) {
        tdm_load_2d_f16(lds_offset(&lK[buf ^ 1][0][0]),
                        Kp + (size_t)(kv0 + 64) * DK, DK, 64, DK, DK, 64, 4, 3);
        __builtin_amdgcn_s_wait_tensorcnt(1);   // tile t done, t+1 in flight
      } else {
        __builtin_amdgcn_s_wait_tensorcnt(0);
      }
    }
    __syncthreads();

    // S = Q * K^T   (four 16-wide n-tiles over the 64 kv columns)
    v8f s[4] = {};
#pragma unroll
    for (int nt = 0; nt < 4; ++nt) {
#pragma unroll
      for (int f = 0; f < 2; ++f) {
        v16h bk = load_bt_frag(&lK[buf][0][0], nt * 16, f * 32, 72);
        s[nt] = wmma_f16(qa[f], bk, s[nt]);
      }
    }

    // scale + causal mask + online softmax
#pragma unroll
    for (int v = 0; v < 8; ++v) {
      const int qrow = q0 + v + 8 * hlf;
      float mx = mrow[v];
#pragma unroll
      for (int nt = 0; nt < 4; ++nt) {
        const int kv = kv0 + nt * 16 + nl;
        float val = (kv <= qrow) ? s[nt][v] * scale : -INFINITY;
        s[nt][v] = val;
        mx = fmaxf(mx, val);
      }
#pragma unroll
      for (int off = 1; off < 16; off <<= 1)
        mx = fmaxf(mx, __shfl_xor(mx, off, 32));

      float alpha = (mrow[v] == -INFINITY) ? 0.f : __expf(mrow[v] - mx);
      float rs = 0.f;
#pragma unroll
      for (int nt = 0; nt < 4; ++nt) {
        float p = (s[nt][v] == -INFINITY) ? 0.f : __expf(s[nt][v] - mx);
        s[nt][v] = p;
        rs += p;
      }
#pragma unroll
      for (int off = 1; off < 16; off <<= 1)
        rs += __shfl_xor(rs, off, 32);

      lrow[v] = lrow[v] * alpha + rs;
      mrow[v] = mx;
#pragma unroll
      for (int nt = 0; nt < 4; ++nt) acc[nt][v] *= alpha;
    }

    // P (C layout) -> A layout via per-wave LDS patch (16x64)
#pragma unroll
    for (int nt = 0; nt < 4; ++nt)
#pragma unroll
      for (int v = 0; v < 8; ++v)
        lP[wave][v + 8 * hlf][nt * 16 + nl] = (_Float16)s[nt][v];

    v16h pa0 = load_a_frag(&lP[wave][0][0], 0, 0,  72);
    v16h pa1 = load_a_frag(&lP[wave][0][0], 0, 32, 72);
#pragma unroll
    for (int nt = 0; nt < 4; ++nt) {
      acc[nt] = wmma_f16(pa0, load_bt_frag(&lVt[0][0], nt * 16, 0,  72), acc[nt]);
      acc[nt] = wmma_f16(pa1, load_bt_frag(&lVt[0][0], nt * 16, 32, 72), acc[nt]);
    }
    __syncthreads();   // protect lVt / lP before next iteration
  }

  // epilogue: 1/l normalize, f16 store to AO[B][S][H*DV]
#pragma unroll
  for (int nt = 0; nt < 4; ++nt) {
#pragma unroll
    for (int v = 0; v < 8; ++v) {
      int m = v + 8 * hlf;
      float val = acc[nt][v] / lrow[v];
      size_t idx = ((size_t)b * SDIM + (q0 + m)) * HV + h * DV + nt * 16 + nl;
      AO[idx] = (_Float16)val;
    }
  }
}

// =====================================================================
// Kernel 3: output projection.  grid = (BS/64, E/64), block = 128.
// K-chunk = 64; f16 A tiles streamed by TDM (double-buffered).
// =====================================================================
__global__ __launch_bounds__(128) void out_proj_kernel(
    const _Float16* __restrict__ AO, const float* __restrict__ Wout,
    const float* __restrict__ bout, float* __restrict__ Out)
{
  const int row0 = blockIdx.x * 64;
  const int n0   = blockIdx.y * 64;
  const int wave = threadIdx.x >> 5;
  const int lane = threadIdx.x & 31;

  __shared__ __align__(16) _Float16 lA[2][64][72];  // [m][k] double-buffer (TDM)
  __shared__ __align__(16) _Float16 lB[64][72];     // [n][k] (W_out layout already)

  v8f acc[4] = {};
  const bool issuer = (threadIdx.x < 32);

  // prime: DMA A tile kc=0 (64 rows x 64 halfs = 32 dwords/row, +4dw pad -> ld 72)
  if (issuer)
    tdm_load_2d_f16(lds_offset(&lA[0][0][0]), &AO[(size_t)row0 * HV],
                    HV, 64, HV, 64, 64, /*interval 32dw=*/4, /*pad 4dw=*/3);

  for (int kc = 0; kc < HV; kc += 64) {
    const int buf = (kc >> 6) & 1;

    // W_out chunk: coalesced float4 along k, b64 stores [n][k]
    for (int i = threadIdx.x; i < (64 * 64) / 4; i += 128) {
      int r = i >> 4, c = (i & 15) * 4;
      v4f w = *(const v4f*)&Wout[(size_t)(n0 + r) * HV + kc + c];
      v4h hw;
#pragma unroll
      for (int j = 0; j < 4; ++j) hw[j] = (_Float16)w[j];
      *(v4h*)&lB[r][c] = hw;
    }

    if (issuer) {
      if (kc + 64 < HV) {
        tdm_load_2d_f16(lds_offset(&lA[buf ^ 1][0][0]),
                        &AO[(size_t)row0 * HV + kc + 64], HV, 64, HV, 64, 64, 4, 3);
        __builtin_amdgcn_s_wait_tensorcnt(1);
      } else {
        __builtin_amdgcn_s_wait_tensorcnt(0);
      }
    }
    __syncthreads();

    v16h a0 = load_a_frag(&lA[buf][0][0], wave * 16, 0,  72);
    v16h a1 = load_a_frag(&lA[buf][0][0], wave * 16, 32, 72);
#pragma unroll
    for (int nt = 0; nt < 4; ++nt) {
      acc[nt] = wmma_f16(a0, load_bt_frag(&lB[0][0], nt * 16, 0,  72), acc[nt]);
      acc[nt] = wmma_f16(a1, load_bt_frag(&lB[0][0], nt * 16, 32, 72), acc[nt]);
    }
    __syncthreads();
  }

  const int hlf = lane >> 4, nl = lane & 15;
#pragma unroll
  for (int nt = 0; nt < 4; ++nt) {
#pragma unroll
    for (int v = 0; v < 8; ++v) {
      int m = v + 8 * hlf;
      int n = n0 + nt * 16 + nl;
      Out[(size_t)(row0 + m) * EDIM + n] = acc[nt][v] + bout[n];
    }
  }
}

// =====================================================================
extern "C" void kernel_launch(void* const* d_in, const int* in_sizes, int n_in,
                              void* d_out, int out_size, void* d_ws, size_t ws_size,
                              hipStream_t stream) {
  const float* X    = (const float*)d_in[0];
  const float* W_Q  = (const float*)d_in[1];
  const float* W_K  = (const float*)d_in[2];
  const float* W_V  = (const float*)d_in[3];
  const float* B_Q  = (const float*)d_in[4];
  const float* B_K  = (const float*)d_in[5];
  const float* B_V  = (const float*)d_in[6];
  const float* Wout = (const float*)d_in[7];
  const float* bout = (const float*)d_in[8];
  float* Out = (float*)d_out;

  const size_t perHead = (size_t)BDIM * HDIM * SDIM * DK;  // 8M f16 elements
  _Float16* Qh = (_Float16*)d_ws;
  _Float16* Kh = Qh + perHead;
  _Float16* Vh = Kh + perHead;
  _Float16* AO = Vh + perHead;   // [B,S,H*DV] — 64 MB of workspace total

  qkv_proj_kernel<<<dim3(BS / 64, 3 * HDIM), 128, 0, stream>>>(
      X, W_Q, W_K, W_V, B_Q, B_K, B_V, Qh, Kh, Vh);

  flash_attn_kernel<<<dim3(SDIM / 64, HDIM, BDIM), 128, 0, stream>>>(Qh, Kh, Vh, AO);

  out_proj_kernel<<<dim3(BS / 64, EDIM / 64), 128, 0, stream>>>(AO, Wout, bout, Out);
}
